// KeywordSpottingModel_with_cls_32658931319053
// MI455X (gfx1250) — matrix operand order repaired and checked by reading, not compiled, not run on hardware
//
#include <hip/hip_runtime.h>
#include <hip/hip_bf16.h>
#include <math.h>

// ---------------- model dims ----------------
#define BATCH   32
#define TIN     200
#define LSEQ    201          // +1 cls token at the end
#define DM      512          // d_model
#define DI      1024         // d_inner
#define DSTATE  16
#define DCONV   4
#define DTR     32           // dt_rank
#define NCLS    12
#define MROWS   (BATCH * LSEQ)   // 6432 = 201 * 32

typedef __attribute__((ext_vector_type(16))) __bf16 v16bf;
typedef __attribute__((ext_vector_type(8)))  __bf16 v8bf;
typedef __attribute__((ext_vector_type(8)))  float  v8f;
typedef __attribute__((ext_vector_type(4)))  unsigned v4u;
typedef __attribute__((ext_vector_type(8)))  int    v8i;
typedef __attribute__((ext_vector_type(4)))  int    v4i;

#if __has_builtin(__builtin_amdgcn_tensor_load_to_lds)
#define HAVE_TDM 1
#else
#define HAVE_TDM 0
#endif

__device__ __forceinline__ __bf16 f32_to_bf16(float f) {
  unsigned u = __builtin_bit_cast(unsigned, f);
  unsigned r = u + 0x7FFFu + ((u >> 16) & 1u);   // round-to-nearest-even
  unsigned short h = (unsigned short)(r >> 16);
  return __builtin_bit_cast(__bf16, h);
}

__device__ __forceinline__ float bf16_to_f32(__bf16 b) {
  unsigned short h = __builtin_bit_cast(unsigned short, b);
  unsigned u = ((unsigned)h) << 16;
  return __builtin_bit_cast(float, u);
}

__device__ __forceinline__ float siluf(float x) {
  return x / (1.0f + expf(-x));
}

// Build v16bf from two contiguous 8-element (16B) runs
__device__ __forceinline__ v16bf frag_from_runs(const __bf16* p0, const __bf16* p1) {
  union { v16bf v; v8bf h[2]; } u;
  u.h[0] = *(const v8bf*)p0;
  u.h[1] = *(const v8bf*)p1;
  return u.v;
}

// 32-bit LDS byte offset of a generic pointer into __shared__ space
typedef __attribute__((address_space(3))) char lds_char_t;
__device__ __forceinline__ unsigned lds_off_u32(void* p) {
  return (unsigned)(unsigned long long)(lds_char_t*)p;
}

#if HAVE_TDM
// Issue one TDM 2D tile load: rows x rowlen (bf16 elems), row stride (elems).
// D# packing per CDNA5 ISA section 8.3/8.4; data_size=1 (2 bytes), no pad,
// no multicast, tensor dims large to disable OOB zeroing.
__device__ __forceinline__ void tdm_load_2d(unsigned lds_addr, const void* gptr,
                                            unsigned rows, unsigned rowlen,
                                            unsigned stride_elems) {
  unsigned long long ga = (unsigned long long)gptr;
  v4u g0;
  g0[0] = 1u;                                            // count=1 (valid)
  g0[1] = lds_addr;                                      // LDS byte address
  g0[2] = (unsigned)(ga & 0xFFFFFFFFu);                  // global_addr lo
  g0[3] = (unsigned)((ga >> 32) & 0x1FFFFFFu) | (2u << 30); // hi | type=2
  const unsigned td0 = 0x7FFFFFFFu, td1 = 0x7FFFFFFFu;
  v8i g1;
  g1[0] = (int)(1u << 16);                               // data_size=1 (2B)
  g1[1] = (int)((td0 & 0xFFFFu) << 16);                  // tensor_dim0 lo16
  g1[2] = (int)(((td0 >> 16) & 0xFFFFu) | ((td1 & 0xFFFFu) << 16));
  g1[3] = (int)(((td1 >> 16) & 0xFFFFu) | (rowlen << 16)); // tile_dim0
  g1[4] = (int)(rows & 0xFFFFu);                         // tile_dim1 (dim2=0)
  g1[5] = (int)stride_elems;                             // dim0_stride lo32
  g1[6] = 0;
  g1[7] = 0;
  v4i g2 = {0, 0, 0, 0};
  v4i g3 = {0, 0, 0, 0};
#if defined(__clang_major__) && (__clang_major__ >= 23)
  __builtin_amdgcn_tensor_load_to_lds(g0, g1, g2, g3, (v8i){0,0,0,0,0,0,0,0}, 0);
#else
  __builtin_amdgcn_tensor_load_to_lds(g0, g1, g2, g3, 0);
#endif
}
#else
// Fallback: cooperative copy of a rows x 32 bf16 tile into LDS
__device__ __forceinline__ void copy_tile(__bf16* dst, const __bf16* src,
                                          int rows, int stride_elems, int tid) {
  const int chunks = rows * 4;   // 4 x 16B chunks per 32-elem row
  for (int c = tid; c < chunks; c += 256) {
    const int r = c >> 2, o = (c & 3) * 8;
    *(v8bf*)(dst + r * 32 + o) = *(const v8bf*)(src + (size_t)r * stride_elems + o);
  }
}
#endif

// ---------------------------------------------------------------------------
// LDS/TDM double-buffered bf16 WMMA GEMM:  C[M,N] = A[M,K(lda)] * W[N,K]^T
// Block: 256 thr (8 waves), block tile 32x256, wave tile 32x32, K step 32.
// grid = (M/32, N/256).  EPI=0 plain store, EPI=1 softplus(v + bias[n]).
// ---------------------------------------------------------------------------
template <int EPI>
__global__ void __launch_bounds__(256)
wmma_gemm_lds_kernel(const __bf16* __restrict__ A, int lda,
                     const __bf16* __restrict__ W,
                     const float* __restrict__ bias,
                     float* __restrict__ C, int ldc, int K) {
  __shared__ __align__(16) __bf16 smA[2][32 * 32];
  __shared__ __align__(16) __bf16 smB[2][256 * 32];

  const int lane = threadIdx.x & 31;
  const int wave = threadIdx.x >> 5;
  const int bm = blockIdx.x;          // 32-row block
  const int bn = blockIdx.y;          // 256-col block
  const int half = lane >> 4;
  const int l15  = lane & 15;

  const __bf16* Ablk = A + (size_t)(bm * 32) * lda;
  const __bf16* Wblk = W + (size_t)(bn * 256) * K;
  const int NK = K >> 5;

#if HAVE_TDM
  if (wave == 0) {
    tdm_load_2d(lds_off_u32(&smA[0][0]), Ablk, 32, 32, (unsigned)lda);
    tdm_load_2d(lds_off_u32(&smB[0][0]), Wblk, 256, 32, (unsigned)K);
  }
#else
  copy_tile(&smA[0][0], Ablk, 32, lda, threadIdx.x);
  copy_tile(&smB[0][0], Wblk, 256, K, threadIdx.x);
#endif

  v8f acc[2][2];
#pragma unroll
  for (int mt = 0; mt < 2; ++mt)
#pragma unroll
    for (int nt = 0; nt < 2; ++nt) acc[mt][nt] = (v8f){};

  for (int kc = 0; kc < NK; ++kc) {
#if HAVE_TDM
    if (wave == 0) __builtin_amdgcn_s_wait_tensorcnt(0);
#endif
    __syncthreads();                       // current buffer ready for all
    const int cur = kc & 1;
    if (kc + 1 < NK) {
      const int k1 = (kc + 1) * 32;
#if HAVE_TDM
      if (wave == 0) {
        tdm_load_2d(lds_off_u32(&smA[cur ^ 1][0]), Ablk + k1, 32, 32, (unsigned)lda);
        tdm_load_2d(lds_off_u32(&smB[cur ^ 1][0]), Wblk + k1, 256, 32, (unsigned)K);
      }
#else
      copy_tile(&smA[cur ^ 1][0], Ablk + k1, 32, lda, threadIdx.x);
      copy_tile(&smB[cur ^ 1][0], Wblk + k1, 256, K, threadIdx.x);
#endif
    }

    const __bf16* sa = &smA[cur][0];
    const __bf16* sb = &smB[cur][0];
    // A fragments (16-bit A 16x32 layout) from LDS
    const int aoff = half * 8;
    const v16bf a0 = frag_from_runs(sa + l15 * 32 + aoff,
                                    sa + l15 * 32 + aoff + 16);
    const v16bf a1 = frag_from_runs(sa + (16 + l15) * 32 + aoff,
                                    sa + (16 + l15) * 32 + aoff + 16);
#pragma unroll
    for (int nt = 0; nt < 2; ++nt) {
      const int n_local = wave * 32 + nt * 16 + l15;
      const __bf16* bp = sb + n_local * 32 + half * 8;
      const v16bf b = frag_from_runs(bp, bp + 16);
      acc[0][nt] = __builtin_amdgcn_wmma_f32_16x16x32_bf16(
          false, a0, false, b, (short)0, acc[0][nt], false, false);
      acc[1][nt] = __builtin_amdgcn_wmma_f32_16x16x32_bf16(
          false, a1, false, b, (short)0, acc[1][nt], false, false);
    }
    __syncthreads();                       // all reads done before overwrite
  }

  // C/D layout: VGPR r -> row r + 8*half, column = lane&15
#pragma unroll
  for (int mt = 0; mt < 2; ++mt) {
#pragma unroll
    for (int nt = 0; nt < 2; ++nt) {
      const int cn = bn * 256 + wave * 32 + nt * 16 + l15;
#pragma unroll
      for (int r = 0; r < 8; ++r) {
        const int cm = bm * 32 + mt * 16 + r + half * 8;
        float v = acc[mt][nt][r];
        if (EPI == 1) {
          v += bias[cn];
          v = (v > 20.0f) ? v : log1pf(expf(v));   // softplus
        }
        C[(size_t)cm * ldc + cn] = v;
      }
    }
  }
}

// ---------------------------------------------------------------------------
// Register-direct bf16 WMMA GEMM (for the small N=64 x_proj GEMM).
// Wave computes 32x64; dual f32 + bf16 store.
// ---------------------------------------------------------------------------
__global__ void wmma_gemm_reg_kernel(const __bf16* __restrict__ A, int lda,
                                     const __bf16* __restrict__ W,
                                     float* __restrict__ C,
                                     __bf16* __restrict__ Cbf, int ldc,
                                     int M, int N, int K) {
  const int lane = threadIdx.x & 31;
  const int wave = threadIdx.x >> 5;
  const int superN = N >> 6;
  const int superTot = (M >> 5) * superN;
  const int sid = blockIdx.x * 8 + wave;     // wave-uniform
  if (sid >= superTot) return;

  const int sm = sid / superN;
  const int sn = sid % superN;
  const int half = lane >> 4;
  const int l15  = lane & 15;

  const __bf16* arow0 = A + (size_t)(sm * 32 + l15) * lda;
  const __bf16* arow1 = arow0 + (size_t)16 * lda;
  const __bf16* wrow0 = W + (size_t)(sn * 64 + l15) * K;
  const __bf16* wrow1 = wrow0 + (size_t)16 * K;
  const __bf16* wrow2 = wrow0 + (size_t)32 * K;
  const __bf16* wrow3 = wrow0 + (size_t)48 * K;

  v8f acc[2][4];
#pragma unroll
  for (int mt = 0; mt < 2; ++mt)
#pragma unroll
    for (int nt = 0; nt < 4; ++nt) acc[mt][nt] = (v8f){};

  for (int k0 = 0; k0 < K; k0 += 32) {
    const int ka = k0 + half * 8;
    const v16bf a0 = frag_from_runs(arow0 + ka, arow0 + ka + 16);
    const v16bf a1 = frag_from_runs(arow1 + ka, arow1 + ka + 16);
    const int kb = k0 + half * 8;
    const __bf16* wr[4] = {wrow0, wrow1, wrow2, wrow3};
#pragma unroll
    for (int nt = 0; nt < 4; ++nt) {
      const v16bf b = frag_from_runs(wr[nt] + kb, wr[nt] + kb + 16);
      acc[0][nt] = __builtin_amdgcn_wmma_f32_16x16x32_bf16(
          false, a0, false, b, (short)0, acc[0][nt], false, false);
      acc[1][nt] = __builtin_amdgcn_wmma_f32_16x16x32_bf16(
          false, a1, false, b, (short)0, acc[1][nt], false, false);
    }
  }

#pragma unroll
  for (int mt = 0; mt < 2; ++mt) {
#pragma unroll
    for (int nt = 0; nt < 4; ++nt) {
      const int cn = sn * 64 + nt * 16 + l15;
#pragma unroll
      for (int r = 0; r < 8; ++r) {
        const int cm = sm * 32 + mt * 16 + r + half * 8;
        const float v = acc[mt][nt][r];
        const size_t off = (size_t)cm * ldc + cn;
        C[off] = v;
        Cbf[off] = f32_to_bf16(v);
      }
    }
  }
}

// ---------------------------------------------------------------------------
__global__ void cvt_bf16_kernel(const float* __restrict__ in,
                                __bf16* __restrict__ out, int n) {
  const int idx = blockIdx.x * blockDim.x + threadIdx.x;
  if (idx < n) out[idx] = f32_to_bf16(in[idx]);
}

// ---------------------------------------------------------------------------
__global__ void proj_kernel(const float* __restrict__ x,
                            const float* __restrict__ pw,
                            const float* __restrict__ pb,
                            const float* __restrict__ cls,
                            __bf16* __restrict__ h) {
  const int idx = blockIdx.x * blockDim.x + threadIdx.x;
  if (idx >= MROWS * DM) return;
  const int m  = idx & (DM - 1);
  const int bt = idx / DM;
  const int t  = bt % LSEQ;
  const int b  = bt / LSEQ;
  if (t == TIN) {
    h[idx] = f32_to_bf16(cls[m]);
    return;
  }
  float acc = pb[m];
  const float* xr = x + (size_t)b * 80 * TIN + t;
  const float* wr = pw + (size_t)m * 80;
#pragma unroll 4
  for (int f = 0; f < 80; ++f) acc += xr[(size_t)f * TIN] * wr[f];
  h[idx] = f32_to_bf16(acc);
}

// ---------------------------------------------------------------------------
__global__ void conv_silu_kernel(const float* __restrict__ xz,
                                 const float* __restrict__ cw,
                                 const float* __restrict__ cb,
                                 float* __restrict__ xcf,
                                 __bf16* __restrict__ xcb) {
  const int idx = blockIdx.x * blockDim.x + threadIdx.x;
  if (idx >= MROWS * DI) return;
  const int d  = idx & (DI - 1);
  const int bt = idx / DI;
  const int t  = bt % LSEQ;
  const int b  = bt / LSEQ;
  float acc = cb[d];
#pragma unroll
  for (int j = 0; j < DCONV; ++j) {
    const int tt = t - (DCONV - 1) + j;
    if (tt >= 0)
      acc += xz[((size_t)(b * LSEQ + tt)) * (2 * DI) + d] * cw[d * DCONV + j];
  }
  const float v = siluf(acc);
  xcf[idx] = v;
  xcb[idx] = f32_to_bf16(v);
}

// ---------------------------------------------------------------------------
__global__ void scan_kernel(const float* __restrict__ dt,
                            const float* __restrict__ xc,
                            const float* __restrict__ dbc,
                            const float* __restrict__ xz,
                            const float* __restrict__ A_log,
                            const float* __restrict__ Dp,
                            __bf16* __restrict__ y) {
  const int d = blockIdx.x * blockDim.x + threadIdx.x;
  const int b = blockIdx.y;

  __shared__ float sB[DSTATE];
  __shared__ float sC[DSTATE];

  float Av[DSTATE];
#pragma unroll
  for (int s = 0; s < DSTATE; ++s) Av[s] = -expf(A_log[d * DSTATE + s]);
  const float Dv = Dp[d];

  float st[DSTATE];
#pragma unroll
  for (int s = 0; s < DSTATE; ++s) st[s] = 0.0f;

  for (int t = 0; t < LSEQ; ++t) {
    __syncthreads();
    if (threadIdx.x < 32) {
      const float v =
          dbc[((size_t)(b * LSEQ + t)) * (DTR + 2 * DSTATE) + DTR + threadIdx.x];
      if (threadIdx.x < DSTATE) sB[threadIdx.x] = v;
      else                      sC[threadIdx.x - DSTATE] = v;
    }
    __syncthreads();

    const size_t rowi = (size_t)(b * LSEQ + t);
    const float dtv = dt[rowi * DI + d];
    const float xv  = xc[rowi * DI + d];
    const float dx  = dtv * xv;
    float yv = 0.0f;
#pragma unroll
    for (int s = 0; s < DSTATE; ++s) {
      st[s] = expf(dtv * Av[s]) * st[s] + dx * sB[s];
      yv += st[s] * sC[s];
    }
    yv += xv * Dv;
    const float z = xz[rowi * (2 * DI) + DI + d];
    y[rowi * DI + d] = f32_to_bf16(yv * siluf(z));
  }
}

// ---------------------------------------------------------------------------
__global__ void rmsnorm_kernel(const float* __restrict__ in,
                               const float* __restrict__ w,
                               __bf16* __restrict__ out) {
  const int row = blockIdx.x;
  const int tid = threadIdx.x;
  const float* r = in + (size_t)row * DM;
  const float v0 = r[tid];
  const float v1 = r[tid + 256];

  __shared__ float red[256];
  red[tid] = v0 * v0 + v1 * v1;
  __syncthreads();
  for (int s = 128; s > 0; s >>= 1) {
    if (tid < s) red[tid] += red[tid + s];
    __syncthreads();
  }
  const float scale = rsqrtf(red[0] / (float)DM + 1.1920929e-7f);
  out[(size_t)row * DM + tid]       = f32_to_bf16(v0 * scale * w[tid]);
  out[(size_t)row * DM + tid + 256] = f32_to_bf16(v1 * scale * w[tid + 256]);
}

// ---------------------------------------------------------------------------
__global__ void fc_kernel(const __bf16* __restrict__ h,
                          const float* __restrict__ fw,
                          const float* __restrict__ fb,
                          float* __restrict__ out) {
  const int idx = blockIdx.x * blockDim.x + threadIdx.x;
  if (idx >= BATCH * NCLS) return;
  const int b = idx / NCLS;
  const int c = idx % NCLS;
  const __bf16* r = h + ((size_t)(b * LSEQ + TIN)) * DM;
  float acc = fb[c];
#pragma unroll 4
  for (int k = 0; k < DM; ++k) acc += bf16_to_f32(r[k]) * fw[c * DM + k];
  out[idx] = acc;
}

// ---------------------------------------------------------------------------
static inline int ceil_div(int a, int b) { return (a + b - 1) / b; }

extern "C" void kernel_launch(void* const* d_in, const int* in_sizes, int n_in,
                              void* d_out, int out_size, void* d_ws, size_t ws_size,
                              hipStream_t stream) {
  const float* x         = (const float*)d_in[0];
  const float* proj_w    = (const float*)d_in[1];
  const float* proj_b    = (const float*)d_in[2];
  const float* cls_token = (const float*)d_in[3];
  const float* in_proj_w = (const float*)d_in[4];
  const float* conv_w    = (const float*)d_in[5];
  const float* conv_b    = (const float*)d_in[6];
  const float* x_proj_w  = (const float*)d_in[7];
  const float* dt_proj_w = (const float*)d_in[8];
  const float* dt_proj_b = (const float*)d_in[9];
  const float* A_log     = (const float*)d_in[10];
  const float* D_param   = (const float*)d_in[11];
  const float* out_proj_w= (const float*)d_in[12];
  const float* rms_w     = (const float*)d_in[13];
  const float* fc_w      = (const float*)d_in[14];
  const float* fc_b      = (const float*)d_in[15];
  float* out = (float*)d_out;

  char* p = (char*)d_ws;
  float*  xz   = (float*)p;  p += (size_t)MROWS * 2 * DI * 4;
  float*  xcf  = (float*)p;  p += (size_t)MROWS * DI * 4;
  float*  dbcf = (float*)p;  p += (size_t)MROWS * 64 * 4;
  float*  dtb  = (float*)p;  p += (size_t)MROWS * DI * 4;
  float*  h2   = (float*)p;  p += (size_t)MROWS * DM * 4;
  __bf16* hbf  = (__bf16*)p; p += (size_t)MROWS * DM * 2;
  __bf16* xcb  = (__bf16*)p; p += (size_t)MROWS * DI * 2;
  __bf16* dbcb = (__bf16*)p; p += (size_t)MROWS * 64 * 2;
  __bf16* ybb  = (__bf16*)p; p += (size_t)MROWS * DI * 2;
  __bf16* w_in = (__bf16*)p; p += (size_t)4 * 2 * DI * DM * 2;
  __bf16* w_xp = (__bf16*)p; p += (size_t)4 * (DTR + 2 * DSTATE) * DI * 2;
  __bf16* w_dt = (__bf16*)p; p += (size_t)4 * DI * DTR * 2;
  __bf16* w_out= (__bf16*)p; p += (size_t)4 * DM * DI * 2;

  // 0) one-shot weight conversions to bf16
  {
    const int n_in_w  = 4 * 2 * DI * DM;
    const int n_xp_w  = 4 * (DTR + 2 * DSTATE) * DI;
    const int n_dt_w  = 4 * DI * DTR;
    const int n_out_w = 4 * DM * DI;
    cvt_bf16_kernel<<<ceil_div(n_in_w, 256), 256, 0, stream>>>(in_proj_w, w_in, n_in_w);
    cvt_bf16_kernel<<<ceil_div(n_xp_w, 256), 256, 0, stream>>>(x_proj_w, w_xp, n_xp_w);
    cvt_bf16_kernel<<<ceil_div(n_dt_w, 256), 256, 0, stream>>>(dt_proj_w, w_dt, n_dt_w);
    cvt_bf16_kernel<<<ceil_div(n_out_w, 256), 256, 0, stream>>>(out_proj_w, w_out, n_out_w);
  }

  // 1) input projection + cls concat (bf16 h)
  proj_kernel<<<ceil_div(MROWS * DM, 256), 256, 0, stream>>>(
      x, proj_w, proj_b, cls_token, hbf);

  for (int i = 0; i < 4; ++i) {
    const __bf16* ipw = w_in  + (size_t)i * 2 * DI * DM;
    const float*  cw  = conv_w + (size_t)i * DI * DCONV;
    const float*  cb  = conv_b + (size_t)i * DI;
    const __bf16* xpw = w_xp  + (size_t)i * (DTR + 2 * DSTATE) * DI;
    const __bf16* dpw = w_dt  + (size_t)i * DI * DTR;
    const float*  dpb = dt_proj_b + (size_t)i * DI;
    const float*  Al  = A_log + (size_t)i * DI * DSTATE;
    const float*  Dpp = D_param + (size_t)i * DI;
    const __bf16* opw = w_out + (size_t)i * DM * DI;
    const float*  rw  = rms_w + (size_t)i * DM;

    // xz = h @ in_proj_w^T   (N=2048, K=512): TDM/LDS pipeline
    wmma_gemm_lds_kernel<0><<<dim3(MROWS / 32, (2 * DI) / 256), 256, 0, stream>>>(
        hbf, DM, ipw, nullptr, xz, 2 * DI, DM);

    // xc = silu(conv(x-half of xz) + cb)  (f32 + bf16)
    conv_silu_kernel<<<ceil_div(MROWS * DI, 256), 256, 0, stream>>>(
        xz, cw, cb, xcf, xcb);

    // dbc = xc @ x_proj_w^T   (N=64, K=1024): register path, dual store
    {
      int st = (MROWS / 32) * ((DTR + 2 * DSTATE) / 64);
      wmma_gemm_reg_kernel<<<ceil_div(st, 8), 256, 0, stream>>>(
          xcb, DI, xpw, dbcf, dbcb, DTR + 2 * DSTATE,
          MROWS, DTR + 2 * DSTATE, DI);
    }

    // dt = softplus(dbc[:, :32] @ dt_proj_w^T + b)   (N=1024, K=32)
    wmma_gemm_lds_kernel<1><<<dim3(MROWS / 32, DI / 256), 256, 0, stream>>>(
        dbcb, DTR + 2 * DSTATE, dpw, dpb, dtb, DI, DTR);

    // selective scan (+ *silu(z) epilogue), bf16 y out
    scan_kernel<<<dim3(DI / 256, BATCH), 256, 0, stream>>>(
        dtb, xcf, dbcf, xz, Al, Dpp, ybb);

    // h2 = y @ out_proj_w^T   (N=512, K=1024)
    wmma_gemm_lds_kernel<0><<<dim3(MROWS / 32, DM / 256), 256, 0, stream>>>(
        ybb, DI, opw, nullptr, h2, DM, DI);

    // h = rmsnorm(h2) * rms_w  (bf16)
    rmsnorm_kernel<<<MROWS, 256, 0, stream>>>(h2, rw, hbf);
  }

  fc_kernel<<<ceil_div(BATCH * NCLS, 128), 128, 0, stream>>>(
      hbf, fc_w, fc_b, out);
}
